// MoEFFNBlock_10402410791099
// MI455X (gfx1250) — matrix operand order repaired and checked by reading, not compiled, not run on hardware
//
#include <hip/hip_runtime.h>

typedef __attribute__((ext_vector_type(16))) __bf16        v16bf;
typedef __attribute__((ext_vector_type(8)))  float         v8f;
typedef __attribute__((ext_vector_type(8)))  unsigned int  v8u;

namespace {

constexpr int TB = 2, TS = 2048, TD = 1024, TH = 2048, TE = 8;
constexpr int T      = TB * TS;      // 4096 tokens
constexpr int NROWS  = 2 * T + T;    // top-2 assignment rows + shared-expert rows = 12288
constexpr int SA     = 40;           // LDS row stride (bf16 elements) for 32-wide K tiles

struct Assign { int i0, i1; float w0, w1; };

__device__ __forceinline__ unsigned short f2bf(float f) {
  unsigned int u = __float_as_uint(f);
  u += 0x7FFFu + ((u >> 16) & 1u);
  return (unsigned short)(u >> 16);
}

__device__ __forceinline__ v16bf pack_frag(uint4 lo, uint4 hi) {
  v8u u;
  u[0] = lo.x; u[1] = lo.y; u[2] = lo.z; u[3] = lo.w;
  u[4] = hi.x; u[5] = hi.y; u[6] = hi.z; u[7] = hi.w;
  return __builtin_bit_cast(v16bf, u);
}

__device__ __forceinline__ v8f wmma_bf16(v16bf a, v16bf b, v8f c) {
  // D(f32 16x16) = A(bf16 16x32) * B(bf16 32x16) + C
  return __builtin_amdgcn_wmma_f32_16x16x32_bf16(false, a, false, b, (short)0, c,
                                                 false, false);
}

// A fragment: lane holds row m=lane&15; K runs [h*8, h*8+8) and [16+h*8, ...), h=lane>>4.
__device__ __forceinline__ v16bf load_fragA(const unsigned short* lds, int row, int lh) {
  const unsigned short* p = lds + row * SA + lh * 8;
  uint4 lo = *reinterpret_cast<const uint4*>(p);
  uint4 hi = *reinterpret_cast<const uint4*>(p + 16);
  return pack_frag(lo, hi);
}

// B fragment: lane holds column n=lane&15; K runs [h*8, h*8+8) and [16+h*8, ...), h=lane>>4.
__device__ __forceinline__ v16bf load_fragB(const unsigned short* lds, int col, int lh) {
  const unsigned short* p = lds + col * SA + lh * 8;
  uint4 lo = *reinterpret_cast<const uint4*>(p);
  uint4 hi = *reinterpret_cast<const uint4*>(p + 16);
  return pack_frag(lo, hi);
}

// gfx1250 async global->LDS copy (ASYNCcnt-tracked, no VGPR round trip).
// Flat LDS addresses carry the wave-relative LDS offset in addr[31:0] (ISA 10.2),
// so the low dword of the generic pointer is the VDST LDS byte address.
__device__ __forceinline__ void async_b128(const unsigned short* g,
                                           const unsigned short* l) {
  unsigned loff = (unsigned)(unsigned long long)(const void*)l;
  asm volatile("global_load_async_to_lds_b128 %0, %1, off"
               :: "v"(loff), "v"(g)
               : "memory");
}

#define WAIT_ASYNC(n) asm volatile("s_wait_asynccnt " #n ::: "memory")

__device__ __forceinline__ int imin(int a, int b) { return a < b ? a : b; }

// ---------------- fp32 -> bf16 bulk conversion ----------------
__global__ void k_cvt(const float4* __restrict__ src, ushort4* __restrict__ dst, int n4) {
  int stride = gridDim.x * blockDim.x;
  for (int i = blockIdx.x * blockDim.x + threadIdx.x; i < n4; i += stride) {
    float4 v = src[i];
    ushort4 o;
    o.x = f2bf(v.x); o.y = f2bf(v.y); o.z = f2bf(v.z); o.w = f2bf(v.w);
    dst[i] = o;
  }
}

// ---------------- router: one wave32 per token ----------------
__global__ void k_router(const float* __restrict__ x, const float* __restrict__ wr,
                         Assign* __restrict__ asg) {
  int lane = threadIdx.x & 31;
  int wave = threadIdx.x >> 5;
  int t = blockIdx.x * (blockDim.x >> 5) + wave;
  if (t >= T) return;
  float acc[TE];
#pragma unroll
  for (int e = 0; e < TE; ++e) acc[e] = 0.f;
  for (int j = lane; j < TD; j += 32) {
    float xv = x[(size_t)t * TD + j];
#pragma unroll
    for (int e = 0; e < TE; ++e) acc[e] += xv * wr[(size_t)e * TD + j];
  }
#pragma unroll
  for (int e = 0; e < TE; ++e)
    for (int off = 16; off > 0; off >>= 1) acc[e] += __shfl_xor(acc[e], off, 32);
  if (lane == 0) {
    int i0 = 0;
#pragma unroll
    for (int e = 1; e < TE; ++e) if (acc[e] > acc[i0]) i0 = e;
    int i1 = (i0 == 0) ? 1 : 0;
#pragma unroll
    for (int e = 0; e < TE; ++e) if (e != i0 && acc[e] > acc[i1]) i1 = e;
    // softmax + top-2 renorm collapses to a 2-way sigmoid of the logit gap
    float w0 = 1.f / (1.f + __expf(acc[i1] - acc[i0]));
    Assign a; a.i0 = i0; a.i1 = i1; a.w0 = w0; a.w1 = 1.f - w0;
    asg[t] = a;
  }
}

// ---------------- expert counts + prefix (single block) ----------------
__global__ void k_count(const Assign* __restrict__ asg, int* __restrict__ seg) {
  __shared__ int cnt[TE];
  if (threadIdx.x < TE) cnt[threadIdx.x] = 0;
  __syncthreads();
  for (int t = threadIdx.x; t < T; t += blockDim.x) {
    Assign a = asg[t];
    atomicAdd(&cnt[a.i0], 1);
    atomicAdd(&cnt[a.i1], 1);
  }
  __syncthreads();
  if (threadIdx.x == 0) {
    int run = 0;
    for (int e = 0; e < TE; ++e) { seg[e] = run; run += cnt[e]; }
    seg[TE]     = run;       // shared "expert 8" segment: all tokens
    seg[TE + 1] = run + T;
  }
}

// ---------------- deterministic ordered compaction ----------------
__global__ void k_scatter(const Assign* __restrict__ asg, const int* __restrict__ seg,
                          int* __restrict__ tok, float* __restrict__ wgt) {
  int e = blockIdx.x;
  if (threadIdx.x != 0) return;
  int pos = seg[e];
  if (e == TE) {
    for (int t = 0; t < T; ++t) { tok[pos] = t; wgt[pos] = 1.f; ++pos; }
  } else {
    for (int t = 0; t < T; ++t) {
      Assign a = asg[t];
      if (a.i0 == e)      { tok[pos] = t; wgt[pos] = a.w0; ++pos; }
      else if (a.i1 == e) { tok[pos] = t; wgt[pos] = a.w1; ++pos; }
    }
  }
}

// ---------------- fused gate/up GEMM + SwiGLU ----------------
// Block tile: 128 rows(tokens) x 64 cols(H). 8 waves in 4x2; wave = 32x32.
// Double-buffered LDS fed by async global->LDS copies (4 issues/wave/tile).
__global__ __launch_bounds__(256)
void k_gateup(const unsigned short* __restrict__ xb,
              const unsigned short* __restrict__ Wg,
              const unsigned short* __restrict__ Wu,
              const unsigned short* __restrict__ sg,
              const unsigned short* __restrict__ su,
              const int* __restrict__ tok, const int* __restrict__ seg,
              unsigned short* __restrict__ hbuf) {
  int e       = blockIdx.z;
  int segBase = seg[e];
  int rows    = seg[e + 1] - segBase;
  int mBase   = blockIdx.x * 128;
  if (mBase >= rows) return;
  int nBase   = blockIdx.y * 64;
  const unsigned short* Bg = (e < TE) ? (Wg + (size_t)e * TH * TD) : sg;
  const unsigned short* Bu = (e < TE) ? (Wu + (size_t)e * TH * TD) : su;

  __shared__ __align__(16) unsigned short lA[2][128 * SA];
  __shared__ __align__(16) unsigned short lG[2][64 * SA];
  __shared__ __align__(16) unsigned short lU[2][64 * SA];

  int tid = threadIdx.x;
  int lane = tid & 31, w = tid >> 5;
  int wr = w >> 1, wc = w & 1;
  int lh = lane >> 4, ml = lane & 15;

  // Gathered A rows: clamp tail rows to a valid row (their outputs are masked
  // in the epilogue; WMMA row m only feeds D row m) so every lane always issues.
  int rA[2], cA[2], tokc[2];
#pragma unroll
  for (int i = 0; i < 2; ++i) {
    int ch = tid + i * 256;
    rA[i] = ch >> 2;
    cA[i] = (ch & 3) * 8;
    tokc[i] = tok[segBase + imin(mBase + rA[i], rows - 1)];
  }
  int rB = tid >> 2, cB = (tid & 3) * 8;

  auto stage = [&](int k0, int buf) {
#pragma unroll
    for (int i = 0; i < 2; ++i)
      async_b128(xb + (size_t)tokc[i] * TD + k0 + cA[i],
                 &lA[buf][rA[i] * SA + cA[i]]);
    async_b128(Bg + (size_t)(nBase + rB) * TD + k0 + cB, &lG[buf][rB * SA + cB]);
    async_b128(Bu + (size_t)(nBase + rB) * TD + k0 + cB, &lU[buf][rB * SA + cB]);
  };

  v8f accG[2][2] = {};
  v8f accU[2][2] = {};

  constexpr int NK = TD / 32;
  stage(0, 0);
  for (int ki = 0; ki < NK; ++ki) {
    int buf = ki & 1;
    if (ki + 1 < NK) {
      stage((ki + 1) * 32, buf ^ 1);
      WAIT_ASYNC(0x4);   // 4 issues for tile ki+1 outstanding => tile ki landed
    } else {
      WAIT_ASYNC(0x0);
    }
    __syncthreads();

    v16bf a0 = load_fragA(lA[buf], wr * 32 + 0 * 16 + ml, lh);
    v16bf a1 = load_fragA(lA[buf], wr * 32 + 1 * 16 + ml, lh);
    v16bf g0 = load_fragB(lG[buf], wc * 32 + 0 * 16 + ml, lh);
    v16bf g1 = load_fragB(lG[buf], wc * 32 + 1 * 16 + ml, lh);
    v16bf u0 = load_fragB(lU[buf], wc * 32 + 0 * 16 + ml, lh);
    v16bf u1 = load_fragB(lU[buf], wc * 32 + 1 * 16 + ml, lh);

    accG[0][0] = wmma_bf16(a0, g0, accG[0][0]);
    accG[0][1] = wmma_bf16(a0, g1, accG[0][1]);
    accG[1][0] = wmma_bf16(a1, g0, accG[1][0]);
    accG[1][1] = wmma_bf16(a1, g1, accG[1][1]);
    accU[0][0] = wmma_bf16(a0, u0, accU[0][0]);
    accU[0][1] = wmma_bf16(a0, u1, accU[0][1]);
    accU[1][0] = wmma_bf16(a1, u0, accU[1][0]);
    accU[1][1] = wmma_bf16(a1, u1, accU[1][1]);
    __syncthreads();   // all waves done reading buf before it is refilled
  }

  // epilogue: h = silu(g) * u -> bf16 rows in the expert's segment
#pragma unroll
  for (int f = 0; f < 2; ++f)
#pragma unroll
    for (int g = 0; g < 2; ++g)
#pragma unroll
      for (int r = 0; r < 8; ++r) {
        int lm = wr * 32 + f * 16 + lh * 8 + r;
        if (mBase + lm < rows) {
          int col = nBase + wc * 32 + g * 16 + ml;
          float gv = accG[f][g][r], uv = accU[f][g][r];
          float hv = gv / (1.f + __expf(-gv)) * uv;
          hbuf[(size_t)(segBase + mBase + lm) * TH + col] = f2bf(hv);
        }
      }
}

// ---------------- down GEMM + weighted combine ----------------
// Block tile: 128 rows x 128 cols(D). 8 waves in 2x4; wave = 64x32.
__global__ __launch_bounds__(256)
void k_down(const unsigned short* __restrict__ hbuf,
            const unsigned short* __restrict__ Wd,
            const unsigned short* __restrict__ sd,
            const int* __restrict__ tok, const float* __restrict__ wgt,
            const int* __restrict__ seg, float* __restrict__ out) {
  int e       = blockIdx.z;
  int segBase = seg[e];
  int rows    = seg[e + 1] - segBase;
  int mBase   = blockIdx.x * 128;
  if (mBase >= rows) return;
  int nBase   = blockIdx.y * 128;
  const unsigned short* Bw = (e < TE) ? (Wd + (size_t)e * TD * TH) : sd;

  __shared__ __align__(16) unsigned short lA[2][128 * SA];
  __shared__ __align__(16) unsigned short lB[2][128 * SA];

  int tid = threadIdx.x;
  int lane = tid & 31, w = tid >> 5;
  int wr = w >> 2, wc = w & 3;
  int lh = lane >> 4, ml = lane & 15;

  int rS[2], cS[2];
  size_t arow[2];
#pragma unroll
  for (int i = 0; i < 2; ++i) {
    int ch = tid + i * 256;
    rS[i] = ch >> 2;
    cS[i] = (ch & 3) * 8;
    arow[i] = (size_t)(segBase + imin(mBase + rS[i], rows - 1));  // clamped tail
  }

  auto stage = [&](int k0, int buf) {
#pragma unroll
    for (int i = 0; i < 2; ++i) {
      async_b128(hbuf + arow[i] * TH + k0 + cS[i], &lA[buf][rS[i] * SA + cS[i]]);
      async_b128(Bw + (size_t)(nBase + rS[i]) * TH + k0 + cS[i],
                 &lB[buf][rS[i] * SA + cS[i]]);
    }
  };

  v8f acc[4][2] = {};

  constexpr int NK = TH / 32;
  stage(0, 0);
  for (int ki = 0; ki < NK; ++ki) {
    int buf = ki & 1;
    if (ki + 1 < NK) {
      stage((ki + 1) * 32, buf ^ 1);
      WAIT_ASYNC(0x4);
    } else {
      WAIT_ASYNC(0x0);
    }
    __syncthreads();

    v16bf af[4];
#pragma unroll
    for (int f = 0; f < 4; ++f) af[f] = load_fragA(lA[buf], wr * 64 + f * 16 + ml, lh);
    v16bf b0 = load_fragB(lB[buf], wc * 32 + 0 * 16 + ml, lh);
    v16bf b1 = load_fragB(lB[buf], wc * 32 + 1 * 16 + ml, lh);
#pragma unroll
    for (int f = 0; f < 4; ++f) {
      acc[f][0] = wmma_bf16(af[f], b0, acc[f][0]);
      acc[f][1] = wmma_bf16(af[f], b1, acc[f][1]);
    }
    __syncthreads();
  }

#pragma unroll
  for (int f = 0; f < 4; ++f)
#pragma unroll
    for (int g = 0; g < 2; ++g)
#pragma unroll
      for (int r = 0; r < 8; ++r) {
        int lm = wr * 64 + f * 16 + lh * 8 + r;
        if (mBase + lm < rows) {
          int row = segBase + mBase + lm;
          int t   = tok[row];
          float wv = wgt[row];
          int col = nBase + wc * 32 + g * 16 + ml;
          atomicAdd(&out[(size_t)t * TD + col], wv * acc[f][g][r]);
        }
      }
}

} // anonymous namespace

extern "C" void kernel_launch(void* const* d_in, const int* in_sizes, int n_in,
                              void* d_out, int out_size, void* d_ws, size_t ws_size,
                              hipStream_t stream) {
  (void)in_sizes; (void)n_in; (void)out_size; (void)ws_size;
  const float* x   = (const float*)d_in[0];
  const float* wr  = (const float*)d_in[1];
  const float* Wg  = (const float*)d_in[2];
  const float* Wu  = (const float*)d_in[3];
  const float* Wd  = (const float*)d_in[4];
  const float* sg  = (const float*)d_in[5];
  const float* su  = (const float*)d_in[6];
  const float* sd  = (const float*)d_in[7];
  float* out = (float*)d_out;

  char* ws = (char*)d_ws;
  size_t off = 0;
  auto carve = [&](size_t bytes) -> char* {
    char* p = ws + off;
    off += (bytes + 255) & ~(size_t)255;
    return p;
  };
  unsigned short* xb   = (unsigned short*)carve((size_t)T * TD * 2);
  unsigned short* Wgb  = (unsigned short*)carve((size_t)TE * TH * TD * 2);
  unsigned short* Wub  = (unsigned short*)carve((size_t)TE * TH * TD * 2);
  unsigned short* Wdb  = (unsigned short*)carve((size_t)TE * TD * TH * 2);
  unsigned short* sgb  = (unsigned short*)carve((size_t)TH * TD * 2);
  unsigned short* sub  = (unsigned short*)carve((size_t)TH * TD * 2);
  unsigned short* sdb  = (unsigned short*)carve((size_t)TD * TH * 2);
  unsigned short* hbuf = (unsigned short*)carve((size_t)NROWS * TH * 2);
  Assign* asg          = (Assign*)carve((size_t)T * sizeof(Assign));
  int*    tokl         = (int*)carve((size_t)NROWS * 4);
  float*  wgtl         = (float*)carve((size_t)NROWS * 4);
  int*    seg          = (int*)carve((size_t)(TE + 2) * 4);

  auto cvt = [&](const float* s, unsigned short* d, size_t n) {
    k_cvt<<<dim3(2048), dim3(256), 0, stream>>>((const float4*)s, (ushort4*)d,
                                                (int)(n / 4));
  };
  cvt(x,  xb,  (size_t)T * TD);
  cvt(Wg, Wgb, (size_t)TE * TH * TD);
  cvt(Wu, Wub, (size_t)TE * TH * TD);
  cvt(Wd, Wdb, (size_t)TE * TD * TH);
  cvt(sg, sgb, (size_t)TH * TD);
  cvt(su, sub, (size_t)TH * TD);
  cvt(sd, sdb, (size_t)TD * TH);

  k_router <<<dim3(T / 8), dim3(256), 0, stream>>>(x, wr, asg);
  k_count  <<<dim3(1),     dim3(256), 0, stream>>>(asg, seg);
  k_scatter<<<dim3(TE + 1), dim3(32), 0, stream>>>(asg, seg, tokl, wgtl);

  hipMemsetAsync(d_out, 0, (size_t)T * TD * sizeof(float), stream);

  dim3 g1(T / 128, TH / 64, TE + 1);   // (32, 32, 9)
  k_gateup<<<g1, 256, 0, stream>>>(xb, Wgb, Wub, sgb, sub, tokl, seg, hbuf);

  dim3 g2(T / 128, TD / 128, TE + 1);  // (32, 8, 9)
  k_down<<<g2, 256, 0, stream>>>(hbuf, Wdb, sdb, tokl, wgtl, seg, out);

  (void)out;
}
